// PipelineMoEBlock_16209206575617
// MI455X (gfx1250) — matrix operand order repaired and checked by reading, not compiled, not run on hardware
//
#include <hip/hip_runtime.h>
#include <math.h>

typedef unsigned short u16;
typedef unsigned int   u32;
typedef unsigned long long u64;
typedef __attribute__((ext_vector_type(16))) __bf16 v16bf;
typedef __attribute__((ext_vector_type(8)))  float  v8f;
typedef __attribute__((ext_vector_type(4)))  unsigned int vu4;
typedef __attribute__((ext_vector_type(8)))  int vi8;
typedef __attribute__((ext_vector_type(4)))  int vi4;

#define BB 8
#define SS 1024
#define DD 1024
#define HH 16
#define HD 64
#define PP 4096
#define EE 8

__device__ __forceinline__ u16 f2bf(float f) {
  u32 u = __float_as_uint(f);
  u32 r = (u + 0x7FFFu + ((u >> 16) & 1u)) >> 16;
  return (u16)r;
}

#if __has_builtin(__builtin_amdgcn_cvt_pk_bf16_f32)
__device__ __forceinline__ u32 pack2bf(float a, float b) {
  auto p = __builtin_amdgcn_cvt_pk_bf16_f32(a, b);
  u32 r;
  __builtin_memcpy(&r, &p, 4);
  return r;
}
#else
__device__ __forceinline__ u32 pack2bf(float a, float b) {
  return (u32)f2bf(a) | ((u32)f2bf(b) << 16);
}
#endif
__device__ __forceinline__ u16 f2bf1(float a) { return (u16)pack2bf(a, a); }

union AFrag { v16bf v; u32 u[8]; u16 s[16]; };

__device__ __forceinline__ v8f zero_v8f() {
  v8f r;
#pragma unroll
  for (int i = 0; i < 8; i++) r[i] = 0.0f;
  return r;
}

// ---------------- Tensor Data Mover: 2D bf16 tile, global -> LDS with row padding ----------------
#if __has_builtin(__builtin_amdgcn_tensor_load_to_lds) && __has_builtin(__builtin_amdgcn_s_wait_tensorcnt)
#define HAVE_TDM 1
#else
#define HAVE_TDM 0
#endif

#if HAVE_TDM
// tile_w elements/row (bf16), tile_h rows, row_stride in elements; LDS gets rows padded by
// pad_amount_enc (+1)*4 bytes every 2^(pad_interval+1) DWORDs.  D# per ISA 8.3/8.4.
__device__ __forceinline__ void tdm_load_2d_bf16(const void* gsrc, u32 lds_byte_off,
                                                 int tile_w, int tile_h,
                                                 long row_stride_elems,
                                                 int pad_interval, int pad_amount_enc) {
  u64 ga = (u64)(uintptr_t)gsrc;
  u64 s0 = (u64)row_stride_elems;
  vu4 g0 = { 0x1u,                                     // count=1, user desc
             lds_byte_off,                             // LDS byte address
             (u32)ga,                                  // global addr [31:0]
             (u32)((ga >> 32) & 0x01FFFFFFull) | 0x80000000u };  // addr[56:32] | type=2
  vi8 g1;
  // w0: data_size=1(2B)<<16 | pad_enable<<20 | pad_interval<<22 | pad_amount<<25
  g1[0] = (int)((1u << 16) | (1u << 20) | ((u32)pad_interval << 22) | ((u32)pad_amount_enc << 25));
  g1[1] = 0;                                           // bar addr=0, tensor_dim0=0x40000000 (lo16=0)
  g1[2] = (int)0x4000u;                                // td0 hi16 | td1(0x40000000) lo16
  g1[3] = (int)(0x4000u | ((u32)tile_w << 16));        // td1 hi16 | tile_dim0
  g1[4] = (int)((u32)tile_h & 0xFFFFu);                // tile_dim1 | tile_dim2=0
  g1[5] = (int)(u32)s0;                                // tensor_dim0_stride[31:0]
  g1[6] = (int)((u32)(s0 >> 32) & 0xFFFFu);            // stride[47:32] | dim1_stride lo=0
  g1[7] = 0;
  vi4 gz = {0, 0, 0, 0};
#if __clang_major__ >= 23
  vi8 gz8 = {0, 0, 0, 0, 0, 0, 0, 0};
  __builtin_amdgcn_tensor_load_to_lds(g0, g1, gz, gz, gz8, 0);
#else
  __builtin_amdgcn_tensor_load_to_lds(g0, g1, gz, gz, 0);
#endif
}
#endif

// ---------------- LayerNorm (fp32 -> bf16), one token per block ----------------
__global__ __launch_bounds__(256)
void ln_bf16_kernel(const float* __restrict__ x, const float* __restrict__ g,
                    const float* __restrict__ b, u16* __restrict__ out) {
  __shared__ float red[2][8];
  int tid = threadIdx.x;
  long base = (long)blockIdx.x * DD;
  float4 v = *(const float4*)(x + base + tid * 4);
  float s = v.x + v.y + v.z + v.w;
  float q = v.x * v.x + v.y * v.y + v.z * v.z + v.w * v.w;
#pragma unroll
  for (int m = 1; m < 32; m <<= 1) {
    s += __shfl_xor(s, m, 32);
    q += __shfl_xor(q, m, 32);
  }
  int w = tid >> 5;
  if ((tid & 31) == 0) { red[0][w] = s; red[1][w] = q; }
  __syncthreads();
  float ts = 0.f, tq = 0.f;
#pragma unroll
  for (int i = 0; i < 8; i++) { ts += red[0][i]; tq += red[1][i]; }
  float mean = ts * (1.0f / DD);
  float var  = tq * (1.0f / DD) - mean * mean;
  float rs   = rsqrtf(var + 1e-5f);
  int c = tid * 4;
  float4 gg = *(const float4*)(g + c);
  float4 bb = *(const float4*)(b + c);
  float a0 = (v.x - mean) * rs * gg.x + bb.x;
  float a1 = (v.y - mean) * rs * gg.y + bb.y;
  float a2 = (v.z - mean) * rs * gg.z + bb.z;
  float a3 = (v.w - mean) * rs * gg.w + bb.w;
  *(uint2*)(out + base + c) = make_uint2(pack2bf(a0, a1), pack2bf(a2, a3));
}

// ---------------- bf16 WMMA GEMM: C[M,N] = A[M,K](bf16) * W[N,K](fp32)^T ----------------
// EPI: 0 = store bf16; 1 = +bias, GELU(exact), store bf16; 2 = +bias(optional) + residual, store fp32
#define LDT 40  // LDS row stride in u16 (80 B)

template <int EPI, bool EXPERT>
__global__ __launch_bounds__(256)
void gemm_bf16_kernel(const u16* __restrict__ A, const float* __restrict__ W,
                      const float* __restrict__ bias, const float* __restrict__ resid,
                      float* __restrict__ outF, u16* __restrict__ outH,
                      int M, int N, int K, long wstride, int bstride) {
  __shared__ __align__(16) u16 As[128 * LDT];
  __shared__ __align__(16) u16 Bs[128 * LDT];
  int tid = threadIdx.x;
  int wv = tid >> 5, lane = tid & 31;
  int half = lane >> 4, lr = lane & 15;
  int wm = wv >> 1, wn = wv & 1;
  int row0 = blockIdx.y * 128, col0 = blockIdx.x * 128;
  (void)M;
  const float* Wp = W;
  const float* bp = bias;
  if (EXPERT) {
    int e = (row0 >> 8) & (EE - 1);
    Wp += (long)e * wstride;
    if (bias) bp = bias + (long)e * bstride;
  }
  v8f acc[2][4];
#pragma unroll
  for (int i = 0; i < 2; i++)
#pragma unroll
    for (int j = 0; j < 4; j++) acc[i][j] = zero_v8f();

  for (int k0 = 0; k0 < K; k0 += 32) {
    // ---- stage A tile (128x32 bf16): TDM if available, else vector loads ----
#if HAVE_TDM
    if (wv == 0) {
      // row = 32 elem = 64B = 16 DWORDs -> pad_interval=3 (16 DW); pad 4 DWORDs -> enc 3
      tdm_load_2d_bf16(A + (long)row0 * K + k0, (u32)(uintptr_t)(&As[0]),
                       32, 128, (long)K, 3, 3);
    }
#else
#pragma unroll
    for (int it = 0; it < 2; it++) {
      int idx = it * 256 + tid;
      int r = idx >> 2, c8 = (idx & 3) * 8;
      uint4 d = *(const uint4*)(A + (long)(row0 + r) * K + k0 + c8);
      *(uint4*)(As + r * LDT + c8) = d;
    }
#endif
    // ---- stage W tile (128x32 fp32 -> bf16 in LDS) ----
#pragma unroll
    for (int it = 0; it < 4; it++) {
      int idx = it * 256 + tid;
      int r = idx >> 3, c4 = (idx & 7) * 4;
      float4 d = *(const float4*)(Wp + (long)(col0 + r) * K + k0 + c4);
      *(u32*)(Bs + r * LDT + c4)     = pack2bf(d.x, d.y);
      *(u32*)(Bs + r * LDT + c4 + 2) = pack2bf(d.z, d.w);
    }
    if (k0 + 32 < K) {
      __builtin_prefetch(Wp + (long)(col0 + (tid >> 1)) * K + k0 + 32, 0, 0);
    }
#if HAVE_TDM
    if (wv == 0) __builtin_amdgcn_s_wait_tensorcnt(0);
#endif
    __syncthreads();

    AFrag af[2], bf[4];
#pragma unroll
    for (int mi = 0; mi < 2; mi++) {
      int r = (wm * 32 + mi * 16 + lr) * LDT;
#pragma unroll
      for (int j = 0; j < 8; j++) {
        int k = 2 * j + ((j >> 2) << 3) + (half ? 8 : 0);  // A-frag K map
        af[mi].u[j] = *(const u32*)(As + r + k);
      }
    }
#pragma unroll
    for (int ni = 0; ni < 4; ni++) {
      int r = (wn * 64 + ni * 16 + lr) * LDT;
#pragma unroll
      for (int j = 0; j < 8; j++) {
        int k = 2 * j + ((j >> 2) << 3) + (half ? 8 : 0);  // B-frag K map
        bf[ni].u[j] = *(const u32*)(Bs + r + k);
      }
    }
#pragma unroll
    for (int mi = 0; mi < 2; mi++)
#pragma unroll
      for (int ni = 0; ni < 4; ni++)
        acc[mi][ni] = __builtin_amdgcn_wmma_f32_16x16x32_bf16(
            false, af[mi].v, false, bf[ni].v, (short)0, acc[mi][ni], false, false);
    __syncthreads();
  }

  // epilogue: C layout => row = base + c + 8*half, col = base + (lane&15)
#pragma unroll
  for (int mi = 0; mi < 2; mi++) {
#pragma unroll
    for (int ni = 0; ni < 4; ni++) {
      int col = col0 + wn * 64 + ni * 16 + lr;
      float bv = 0.0f;
      if (EPI >= 1 && bp) bv = bp[col];
#pragma unroll
      for (int c = 0; c < 8; c++) {
        int row = row0 + wm * 32 + mi * 16 + c + (half ? 8 : 0);
        float v = acc[mi][ni][c] + bv;
        long off = (long)row * N + col;
        if (EPI == 0) {
          outH[off] = f2bf1(v);
        } else if (EPI == 1) {
          float gl = 0.5f * v * (1.0f + erff(v * 0.70710678118f));
          outH[off] = f2bf1(gl);
        } else {
          outF[off] = v + resid[off];
        }
      }
    }
  }
}

// ---------------- Flash attention: per block = (b, h, 128 q-rows), 8 waves x 16 rows ----------------
#define KSTR 72
#define VSTR 40
#define PSTR 40

__global__ __launch_bounds__(256)
void attn_kernel(const u16* __restrict__ Q, const u16* __restrict__ Kb,
                 const u16* __restrict__ Vb, u16* __restrict__ O) {
  __shared__ __align__(16) u16 Kt[32 * KSTR];
  __shared__ __align__(16) u16 VtT[64 * VSTR];
  __shared__ __align__(16) u16 Pl[8][16 * PSTR];
  int tid = threadIdx.x;
  int wv = tid >> 5, lane = tid & 31;
  int half = lane >> 4, lr = lane & 15;
  int qt = blockIdx.x, h = blockIdx.y, b = blockIdx.z;
  int qbase = qt * 128 + wv * 16;
  long headoff = (long)h * HD;

  // Q fragments (16x64 -> two 16x32 A-frags) held in registers
  AFrag qf[2];
  {
    const u16* qp = Q + ((long)(b * SS + qbase + lr)) * DD + headoff;
#pragma unroll
    for (int kk = 0; kk < 2; kk++)
#pragma unroll
      for (int j = 0; j < 8; j++) {
        int k = 2 * j + ((j >> 2) << 3) + (half ? 8 : 0) + 32 * kk;
        qf[kk].u[j] = *(const u32*)(qp + k);
      }
  }

  float m_i[8], l_i[8];
#pragma unroll
  for (int c = 0; c < 8; c++) { m_i[c] = -1e30f; l_i[c] = 0.0f; }
  v8f oacc[4];
#pragma unroll
  for (int n = 0; n < 4; n++) oacc[n] = zero_v8f();

  const float scale = 0.125f;  // 1/sqrt(64)

  for (int kc = 0; kc < SS; kc += 32) {
    __syncthreads();  // protect previous iteration's LDS reads
#if HAVE_TDM
    if (wv == 0) {
      // K tile 32x64: row = 128B = 32 DWORDs -> pad_interval=4; pad 4 DWORDs -> enc 3
      tdm_load_2d_bf16(Kb + ((long)(b * SS + kc)) * DD + headoff,
                       (u32)(uintptr_t)(&Kt[0]), 64, 32, (long)DD, 4, 3);
    }
    {
      int r = tid >> 3, c8 = (tid & 7) * 8;
      const u16* vp = Vb + ((long)(b * SS + kc + r)) * DD + headoff + c8;
      union { uint4 q; u16 s[8]; } vd;
      vd.q = *(const uint4*)vp;
#pragma unroll
      for (int t = 0; t < 8; t++) VtT[(c8 + t) * VSTR + r] = vd.s[t];  // transpose V
    }
    if (wv == 0) __builtin_amdgcn_s_wait_tensorcnt(0);
#else
    {
      int r = tid >> 3, c8 = (tid & 7) * 8;
      const u16* kp = Kb + ((long)(b * SS + kc + r)) * DD + headoff + c8;
      *(uint4*)(Kt + r * KSTR + c8) = *(const uint4*)kp;
      const u16* vp = Vb + ((long)(b * SS + kc + r)) * DD + headoff + c8;
      union { uint4 q; u16 s[8]; } vd;
      vd.q = *(const uint4*)vp;
#pragma unroll
      for (int t = 0; t < 8; t++) VtT[(c8 + t) * VSTR + r] = vd.s[t];  // transpose V
    }
#endif
    __syncthreads();

    // scores: S[16q x 32keys] = Q(16x64) * K^T
    v8f st[2];
    st[0] = zero_v8f(); st[1] = zero_v8f();
#pragma unroll
    for (int s = 0; s < 2; s++) {
#pragma unroll
      for (int kk = 0; kk < 2; kk++) {
        AFrag kf;
#pragma unroll
        for (int j = 0; j < 8; j++) {
          int kd = 2 * j + ((j >> 2) << 3) + (half ? 8 : 0) + 32 * kk;
          kf.u[j] = *(const u32*)(Kt + (s * 16 + lr) * KSTR + kd);
        }
        st[s] = __builtin_amdgcn_wmma_f32_16x16x32_bf16(
            false, qf[kk].v, false, kf.v, (short)0, st[s], false, false);
      }
    }

    // online softmax over this 32-key chunk (row reductions across 16-lane halves)
#pragma unroll
    for (int c = 0; c < 8; c++) {
      float s0 = st[0][c] * scale;
      float s1 = st[1][c] * scale;
      float mx = fmaxf(s0, s1);
#pragma unroll
      for (int m = 1; m < 16; m <<= 1) mx = fmaxf(mx, __shfl_xor(mx, m, 32));
      float mnew = fmaxf(m_i[c], mx);
      float corr = __expf(m_i[c] - mnew);
      float p0 = __expf(s0 - mnew);
      float p1 = __expf(s1 - mnew);
      float rs = p0 + p1;
#pragma unroll
      for (int m = 1; m < 16; m <<= 1) rs += __shfl_xor(rs, m, 32);
      l_i[c] = l_i[c] * corr + rs;
      m_i[c] = mnew;
#pragma unroll
      for (int n = 0; n < 4; n++) oacc[n][c] *= corr;
      int prow = c + (half ? 8 : 0);
      Pl[wv][prow * PSTR + lr]      = f2bf1(p0);
      Pl[wv][prow * PSTR + 16 + lr] = f2bf1(p1);
    }
    __syncthreads();

    // O += P(16x32) * V(32x64)
    AFrag pf;
#pragma unroll
    for (int j = 0; j < 8; j++) {
      int k = 2 * j + ((j >> 2) << 3) + (half ? 8 : 0);
      pf.u[j] = *(const u32*)(&Pl[wv][lr * PSTR + k]);
    }
#pragma unroll
    for (int n = 0; n < 4; n++) {
      AFrag vf;
#pragma unroll
      for (int j = 0; j < 8; j++) {
        int k = 2 * j + ((j >> 2) << 3) + (half ? 8 : 0);
        vf.u[j] = *(const u32*)(VtT + (n * 16 + lr) * VSTR + k);
      }
      oacc[n] = __builtin_amdgcn_wmma_f32_16x16x32_bf16(
          false, pf.v, false, vf.v, (short)0, oacc[n], false, false);
    }
  }

  // epilogue: normalize by l and store bf16
#pragma unroll
  for (int n = 0; n < 4; n++)
#pragma unroll
    for (int c = 0; c < 8; c++) {
      int row = qbase + c + (half ? 8 : 0);
      int col = (int)headoff + n * 16 + lr;
      O[((long)(b * SS + row)) * DD + col] = f2bf1(oacc[n][c] / l_i[c]);
    }
}

// ---------------- launch ----------------
extern "C" void kernel_launch(void* const* d_in, const int* in_sizes, int n_in,
                              void* d_out, int out_size, void* d_ws, size_t ws_size,
                              hipStream_t stream) {
  (void)in_sizes; (void)n_in; (void)out_size; (void)ws_size;
  const float* x     = (const float*)d_in[0];
  const float* ln1_g = (const float*)d_in[1];
  const float* ln1_b = (const float*)d_in[2];
  const float* wq    = (const float*)d_in[3];
  const float* wk    = (const float*)d_in[4];
  const float* wvp   = (const float*)d_in[5];
  const float* wo    = (const float*)d_in[6];
  const float* moe_g = (const float*)d_in[7];
  const float* moe_b = (const float*)d_in[8];
  // d_in[9] = gate_w : logits unused downstream in the reference -> skipped
  const float* fc1_w = (const float*)d_in[10];
  const float* fc1_b = (const float*)d_in[11];
  const float* fc2_w = (const float*)d_in[12];
  const float* fc2_b = (const float*)d_in[13];
  float* out = (float*)d_out;

  char* ws = (char*)d_ws;
  const unsigned T = BB * SS;  // 8192 tokens
  const size_t MB1 = 1024u * 1024u;
  u16*   xn    = (u16*)(ws);                 // 16 MB
  u16*   qb    = (u16*)(ws + 16 * MB1);      // 16 MB
  u16*   kb    = (u16*)(ws + 32 * MB1);      // 16 MB
  u16*   vb    = (u16*)(ws + 48 * MB1);      // 16 MB
  u16*   ob    = (u16*)(ws + 64 * MB1);      // 16 MB
  float* xproc = (float*)(ws + 80 * MB1);    // 32 MB
  u16*   xm    = (u16*)(ws + 112 * MB1);     // 16 MB
  u16*   hb    = (u16*)(ws + 128 * MB1);     // 64 MB

  dim3 blk(256);
  dim3 gD(DD / 128, T / 128);

  // LN1 -> xn (bf16)
  ln_bf16_kernel<<<dim3(T), blk, 0, stream>>>(x, ln1_g, ln1_b, xn);
  // Q/K/V projections
  gemm_bf16_kernel<0, false><<<gD, blk, 0, stream>>>(xn, wq,  nullptr, nullptr, nullptr, qb, T, DD, DD, 0, 0);
  gemm_bf16_kernel<0, false><<<gD, blk, 0, stream>>>(xn, wk,  nullptr, nullptr, nullptr, kb, T, DD, DD, 0, 0);
  gemm_bf16_kernel<0, false><<<gD, blk, 0, stream>>>(xn, wvp, nullptr, nullptr, nullptr, vb, T, DD, DD, 0, 0);
  // attention
  attn_kernel<<<dim3(SS / 128, HH, BB), blk, 0, stream>>>(qb, kb, vb, ob);
  // output projection + residual -> x_proc (fp32)
  gemm_bf16_kernel<2, false><<<gD, blk, 0, stream>>>(ob, wo, nullptr, x, xproc, nullptr, T, DD, DD, 0, 0);
  // LN2 -> xm (bf16)
  ln_bf16_kernel<<<dim3(T), blk, 0, stream>>>(xproc, moe_g, moe_b, xm);
  // MoE fc1 (+bias, exact GELU) -> h (bf16)
  gemm_bf16_kernel<1, true><<<dim3(PP / 128, T / 128), blk, 0, stream>>>(
      xm, fc1_w, fc1_b, nullptr, nullptr, hb, T, PP, DD, (long)PP * DD, PP);
  // MoE fc2 (+bias) + residual x_proc -> out (fp32)
  gemm_bf16_kernel<2, true><<<gD, blk, 0, stream>>>(
      hb, fc2_w, fc2_b, xproc, out, nullptr, T, DD, PP, (long)DD * PP, DD);
}
